// Mamba_59158879535150
// MI455X (gfx1250) — compile-verified, matrix operand order checked by reading, not run on hardware
//
#include <hip/hip_runtime.h>
#include <math.h>

// ---- model dims -------------------------------------------------------------
#define V_   32000
#define D_   1024
#define N_   16
#define KC_  4
#define DI_  2048          // E*D
#define R_   64            // (D+15)/16
#define B__  4
#define L_   2048
#define T_   (B__ * L_)    // 8192 tokens

#define LDSW 72            // LDS row stride in bf16 elems (32 dword row + 4 dword pad)
#define TCH  8             // conv timesteps per thread
#define SCH  8             // scan timesteps per chunk

typedef __attribute__((ext_vector_type(16))) __bf16 v16bf;
typedef __attribute__((ext_vector_type(8)))  __bf16 v8bf;
typedef __attribute__((ext_vector_type(4)))  __bf16 v4bf;
typedef __attribute__((ext_vector_type(8)))  float  v8f;
typedef __attribute__((ext_vector_type(4)))  float  v4f;
typedef __attribute__((ext_vector_type(4)))  unsigned int v4u;
typedef __attribute__((ext_vector_type(8)))  int    v8i;
typedef __attribute__((ext_vector_type(4)))  int    v4i;

#if __has_include(<hip/amd_detail/amd_gfx1250_TDM.h>)
#define TDM_AVAILABLE 1
#define TDM_SIX_ARG 1
#elif __has_builtin(__builtin_amdgcn_tensor_load_to_lds)
#define TDM_AVAILABLE 1
#define TDM_SIX_ARG 0
#endif

// ---- helpers ----------------------------------------------------------------
static __device__ __forceinline__ float sigmoidf_(float x) {
  return 1.0f / (1.0f + __expf(-x));
}

// Build one 16x32 bf16 WMMA fragment for an NT-gemm operand (row-major over K).
// ISA 16-bit A layout (B symmetric for NT): lanes 0-15 hold K {0..7,16..23},
// lanes 16-31 hold K {8..15,24..31}.
static __device__ __forceinline__ v16bf
load_frag_nt(const __bf16* __restrict__ base, int ld, int row, int k0, bool hi) {
  const __bf16* p = base + (size_t)row * ld + k0 + (hi ? 8 : 0);
  v8bf lo = *(const v8bf*)p;
  v8bf hh = *(const v8bf*)(p + 16);
  v16bf r;
#pragma unroll
  for (int i = 0; i < 8; ++i) { r[i] = lo[i]; r[8 + i] = hh[i]; }
  return r;
}

// Same fragment build but from a padded LDS tile (row stride LDSW elems).
static __device__ __forceinline__ v16bf
load_frag_lds(const __bf16* base, int row, int kk, bool hi) {
  const __bf16* p = base + row * LDSW + kk + (hi ? 8 : 0);
  v8bf lo = *(const v8bf*)p;       // ds_load_b128, conflict-free (stride 36 dwords)
  v8bf hh = *(const v8bf*)(p + 16);
  v16bf r;
#pragma unroll
  for (int i = 0; i < 8; ++i) { r[i] = lo[i]; r[8 + i] = hh[i]; }
  return r;
}

#define WMMA_BF16(a, b, c) \
  __builtin_amdgcn_wmma_f32_16x16x32_bf16(false, (a), false, (b), (short)0, (c), false, false)

#if defined(TDM_AVAILABLE)
// TDM: load a rows x 64 bf16 tile (global row stride gld elems) into LDS at
// byte offset lds_off, padding 4 dwords after every 32 dwords (row) so the LDS
// row stride is 144B -> ds_load_b128 bank-conflict-free.
static __device__ __forceinline__ void
tdm_fill_tile(unsigned lds_off, const void* gptr, unsigned rows, unsigned gld) {
  unsigned long long ga = (unsigned long long)(size_t)gptr;
  v4u g0;
  g0[0] = 1u;                                             // count=1, user mode
  g0[1] = lds_off;                                        // lds_addr
  g0[2] = (unsigned)ga;                                   // global_addr[31:0]
  g0[3] = (unsigned)((ga >> 32) & 0x01FFFFFFu) | (2u << 30); // addr[56:32] | type=2
  v8i g1;
  g1[0] = (int)((1u << 16)   // data_size = 2B
              | (1u << 20)   // pad_enable
              | (4u << 22)   // pad_interval: 32 dwords
              | (3u << 25)); // pad_amount: 4 dwords
  g1[1] = (int)((64u & 0xFFFFu) << 16);                         // tensor_dim0 lo
  g1[2] = (int)(((64u >> 16) & 0xFFFFu) | ((rows & 0xFFFFu) << 16)); // dim0 hi|dim1 lo
  g1[3] = (int)(((rows >> 16) & 0xFFFFu) | (64u << 16));        // dim1 hi | tile_dim0
  g1[4] = (int)(rows & 0xFFFFu);                                // tile_dim1 (tile_dim2=0)
  g1[5] = (int)gld;                                             // tensor_dim0_stride lo
  g1[6] = 0;
  g1[7] = 0;
  v4i gz = {0, 0, 0, 0};
#if TDM_SIX_ARG
  v8i gz8 = {0, 0, 0, 0, 0, 0, 0, 0};
  __builtin_amdgcn_tensor_load_to_lds(g0, g1, gz, gz, gz8, 0);
#else
  __builtin_amdgcn_tensor_load_to_lds(g0, g1, gz, gz, 0);
#endif
}
#else
// Fallback: cooperative global->LDS copy with the same padded layout.
static __device__ __forceinline__ void
fill_tile_fb(__bf16* sdst, const __bf16* __restrict__ gsrc, int gld, int rows) {
  int n8 = rows * 64 / 8;
  for (int e8 = threadIdx.x; e8 < n8; e8 += 256) {
    int e = e8 * 8;
    int r = e >> 6, c = e & 63;
    v8bf v = *(const v8bf*)(gsrc + (size_t)r * gld + c);
    *(v8bf*)(sdst + r * LDSW + c) = v;
  }
}
#endif

// ---- kernel 1: embedding gather + f32->bf16 cast ----------------------------
__global__ __launch_bounds__(256)
void k_gather_bf16(const int* __restrict__ tokens, const float* __restrict__ emb,
                   __bf16* __restrict__ h_bf) {
  int t  = blockIdx.x;
  int d0 = threadIdx.x * 4;
  int tok = tokens[t];
  const float4 v = *(const float4*)(emb + (size_t)tok * D_ + d0);
  v4bf o;
  o[0] = (__bf16)v.x; o[1] = (__bf16)v.y; o[2] = (__bf16)v.z; o[3] = (__bf16)v.w;
  *(v4bf*)(h_bf + (size_t)t * D_ + d0) = o;
}

// ---- generic f32 -> bf16 cast ----------------------------------------------
__global__ __launch_bounds__(256)
void k_cast_bf16(const float* __restrict__ src, __bf16* __restrict__ dst, int n) {
  int i = blockIdx.x * 256 + threadIdx.x;
  if (i < n) dst[i] = (__bf16)src[i];
}

// ---- GEMM 1: x_pre[T,DI] = h_bf[T,D] @ w_in_x_bf[DI,D]^T --------------------
// TDM double-buffered LDS tiles (A:128x64, B:64x64 per 64-wide k-step);
// 8 waves laid out 4(M)x2(N); wave tile 32x32; block tile 128x64.
__global__ __launch_bounds__(256)
void k_gemm_xpre(const __bf16* __restrict__ A, const __bf16* __restrict__ Bw,
                 float* __restrict__ C) {
  __shared__ __bf16 smem[2 * 128 * LDSW + 2 * 64 * LDSW];
  const int lane   = threadIdx.x & 31;
  const int wave   = threadIdx.x >> 5;
  const int wm     = wave & 3, wn = wave >> 2;
  const int rowB   = blockIdx.x * 128;
  const int colB   = blockIdx.y * 64;
  const int laneLo = lane & 15;
  const bool hi    = lane >= 16;

  __bf16* sA0 = smem;
  __bf16* sB0 = smem + 2 * 128 * LDSW;
#if defined(TDM_AVAILABLE)
  const unsigned lbase = (unsigned)(size_t)(void*)smem;
#define FILL_STAGE(bufi, k0e)                                                        \
  do { if (threadIdx.x < 32) {                                                       \
    tdm_fill_tile(lbase + (unsigned)(bufi) * (128u * LDSW * 2u),                     \
                  A + (size_t)rowB * D_ + (k0e), 128u, D_);                          \
    tdm_fill_tile(lbase + 2u * 128u * LDSW * 2u + (unsigned)(bufi) * (64u * LDSW * 2u), \
                  Bw + (size_t)colB * D_ + (k0e), 64u, D_);                          \
  } } while (0)
#define STAGE_SYNC()                                                                 \
  do { if (threadIdx.x < 32) __builtin_amdgcn_s_wait_tensorcnt(0);                   \
       __syncthreads(); } while (0)
#else
#define FILL_STAGE(bufi, k0e)                                                        \
  do { fill_tile_fb(sA0 + (bufi) * 128 * LDSW, A + (size_t)rowB * D_ + (k0e), D_, 128); \
       fill_tile_fb(sB0 + (bufi) * 64 * LDSW, Bw + (size_t)colB * D_ + (k0e), D_, 64);  \
  } while (0)
#define STAGE_SYNC() __syncthreads()
#endif

  v8f acc[2][2] = {};
  FILL_STAGE(0, 0);
  STAGE_SYNC();
  for (int it = 0; it < D_ / 64; ++it) {
    if (it + 1 < D_ / 64) FILL_STAGE((it + 1) & 1, (it + 1) * 64);
    const __bf16* tA = sA0 + (it & 1) * 128 * LDSW;
    const __bf16* tB = sB0 + (it & 1) * 64 * LDSW;
#pragma unroll
    for (int kk = 0; kk < 64; kk += 32) {
      v16bf a0 = load_frag_lds(tA, wm * 32 + 0  + laneLo, kk, hi);
      v16bf a1 = load_frag_lds(tA, wm * 32 + 16 + laneLo, kk, hi);
      v16bf b0 = load_frag_lds(tB, wn * 32 + 0  + laneLo, kk, hi);
      v16bf b1 = load_frag_lds(tB, wn * 32 + 16 + laneLo, kk, hi);
      acc[0][0] = WMMA_BF16(a0, b0, acc[0][0]);
      acc[0][1] = WMMA_BF16(a0, b1, acc[0][1]);
      acc[1][0] = WMMA_BF16(a1, b0, acc[1][0]);
      acc[1][1] = WMMA_BF16(a1, b1, acc[1][1]);
    }
    STAGE_SYNC();
  }
#pragma unroll
  for (int mt = 0; mt < 2; ++mt)
#pragma unroll
    for (int nt = 0; nt < 2; ++nt) {
      int row0 = rowB + wm * 32 + mt * 16 + (hi ? 8 : 0);
      int col  = colB + wn * 32 + nt * 16 + laneLo;
#pragma unroll
      for (int r = 0; r < 8; ++r)
        C[(size_t)(row0 + r) * DI_ + col] = acc[mt][nt][r];
    }
#undef FILL_STAGE
#undef STAGE_SYNC
}

// ---- conv (depthwise causal K=4) + bias + SiLU; 8 timesteps per thread ------
__global__ __launch_bounds__(256)
void k_conv_silu(const float* __restrict__ xpre, const float* __restrict__ cw,
                 const float* __restrict__ cb, __bf16* __restrict__ x_bf) {
  int g  = blockIdx.x * 256 + threadIdx.x;   // (T_/TCH)*DI_ threads
  int d  = g & (DI_ - 1);
  int tc = g >> 11;                          // / DI_
  int t0 = tc * TCH;                         // chunks never cross batch boundary
  int tl0 = t0 & (L_ - 1);
  float w0 = cw[d * 4 + 0], w1 = cw[d * 4 + 1];
  float w2 = cw[d * 4 + 2], w3 = cw[d * 4 + 3];
  float bias = cb[d];
  float xb[TCH + 3];
#pragma unroll
  for (int i = 0; i < TCH + 3; ++i) {
    int tl = tl0 - 3 + i;
    xb[i] = (tl >= 0) ? xpre[(size_t)(t0 - 3 + i) * DI_ + d] : 0.0f;
  }
#pragma unroll
  for (int j = 0; j < TCH; ++j) {
    float acc = bias + xb[j] * w0 + xb[j + 1] * w1 + xb[j + 2] * w2 + xb[j + 3] * w3;
    float s = acc * sigmoidf_(acc);
    x_bf[(size_t)(t0 + j) * DI_ + d] = (__bf16)s;
  }
}

// ---- GEMM 2: dbc[T,96] = x_bf[T,DI] @ w_x_bf[96,DI]^T; also emit dtr bf16 ---
__global__ __launch_bounds__(256)
void k_gemm_dbc(const __bf16* __restrict__ A, const __bf16* __restrict__ Bw,
                float* __restrict__ C, __bf16* __restrict__ dtr) {
  const int lane   = threadIdx.x & 31;
  const int wave   = threadIdx.x >> 5;
  const int wm     = wave & 3, wn = wave >> 2;
  const int rowB   = blockIdx.x * 128 + wm * 32;
  const int colB   = wn * 48;
  const int laneLo = lane & 15;
  const bool hi    = lane >= 16;

  v8f acc[2][3] = {};
  for (int k0 = 0; k0 < DI_; k0 += 32) {
    v16bf a0 = load_frag_nt(A, DI_, rowB + 0  + laneLo, k0, hi);
    v16bf a1 = load_frag_nt(A, DI_, rowB + 16 + laneLo, k0, hi);
    v16bf b0 = load_frag_nt(Bw, DI_, colB + 0  + laneLo, k0, hi);
    v16bf b1 = load_frag_nt(Bw, DI_, colB + 16 + laneLo, k0, hi);
    v16bf b2 = load_frag_nt(Bw, DI_, colB + 32 + laneLo, k0, hi);
    acc[0][0] = WMMA_BF16(a0, b0, acc[0][0]);
    acc[0][1] = WMMA_BF16(a0, b1, acc[0][1]);
    acc[0][2] = WMMA_BF16(a0, b2, acc[0][2]);
    acc[1][0] = WMMA_BF16(a1, b0, acc[1][0]);
    acc[1][1] = WMMA_BF16(a1, b1, acc[1][1]);
    acc[1][2] = WMMA_BF16(a1, b2, acc[1][2]);
  }
#pragma unroll
  for (int mt = 0; mt < 2; ++mt)
#pragma unroll
    for (int nt = 0; nt < 3; ++nt) {
      int row0 = rowB + mt * 16 + (hi ? 8 : 0);
      int col  = colB + nt * 16 + laneLo;
#pragma unroll
      for (int r = 0; r < 8; ++r) {
        float v = acc[mt][nt][r];
        C[(size_t)(row0 + r) * 96 + col] = v;
        if (col < R_) dtr[(size_t)(row0 + r) * R_ + col] = (__bf16)v;
      }
    }
}

// ---- GEMM 3: dt[T,DI] = softplus(dtr[T,64] @ w_dt_bf[DI,64]^T + b_dt) ------
__global__ __launch_bounds__(256)
void k_gemm_dt(const __bf16* __restrict__ A, const __bf16* __restrict__ Bw,
               const float* __restrict__ b_dt, float* __restrict__ dtW) {
  const int lane   = threadIdx.x & 31;
  const int wave   = threadIdx.x >> 5;
  const int wm     = wave & 3, wn = wave >> 2;
  const int rowB   = blockIdx.x * 128 + wm * 32;
  const int colB   = blockIdx.y * 64  + wn * 32;
  const int laneLo = lane & 15;
  const bool hi    = lane >= 16;

  v8f acc[2][2] = {};
#pragma unroll
  for (int k0 = 0; k0 < R_; k0 += 32) {
    v16bf a0 = load_frag_nt(A,  R_, rowB + 0  + laneLo, k0, hi);
    v16bf a1 = load_frag_nt(A,  R_, rowB + 16 + laneLo, k0, hi);
    v16bf b0 = load_frag_nt(Bw, R_, colB + 0  + laneLo, k0, hi);
    v16bf b1 = load_frag_nt(Bw, R_, colB + 16 + laneLo, k0, hi);
    acc[0][0] = WMMA_BF16(a0, b0, acc[0][0]);
    acc[0][1] = WMMA_BF16(a0, b1, acc[0][1]);
    acc[1][0] = WMMA_BF16(a1, b0, acc[1][0]);
    acc[1][1] = WMMA_BF16(a1, b1, acc[1][1]);
  }
#pragma unroll
  for (int mt = 0; mt < 2; ++mt)
#pragma unroll
    for (int nt = 0; nt < 2; ++nt) {
      int row0 = rowB + mt * 16 + (hi ? 8 : 0);
      int col  = colB + nt * 16 + laneLo;
      float bb = b_dt[col];
#pragma unroll
      for (int r = 0; r < 8; ++r) {
        float u  = acc[mt][nt][r] + bb;
        float sp = (u > 20.0f) ? u : log1pf(__expf(u));
        dtW[(size_t)(row0 + r) * DI_ + col] = sp;
      }
    }
}

// ---- selective scan: one thread per (b,d,n); no barriers, chunked loads -----
// 512 blocks x 256 thr; block = (b, 16 d-channels); thread = (d_local, n).
// Final y = sum_n st*C via __shfl_xor within 16-lane groups (wave32-native).
__global__ __launch_bounds__(256)
void k_scan(const float* __restrict__ dtW, const __bf16* __restrict__ x_bf,
            const float* __restrict__ dbc, const float* __restrict__ A_log,
            const float* __restrict__ Dp, float* __restrict__ y_last) {
  int b  = blockIdx.x >> 7;                 // 512 blocks: b = bid / 128
  int dc = blockIdx.x & 127;                // d-chunk of 16
  int dl = threadIdx.x >> 4;                // local d
  int n  = threadIdx.x & 15;                // state channel
  int d  = dc * 16 + dl;
  float Aval = -__expf(A_log[d * N_ + n]);
  float st = 0.0f;
  const size_t base = (size_t)b * L_;
  float lastC = 0.0f, lastX = 0.0f;
  for (int t0 = 0; t0 < L_; t0 += SCH) {
    float dtv[SCH], xv[SCH], Bv[SCH], Cv[SCH];
#pragma unroll
    for (int j = 0; j < SCH; ++j) {          // batched issue, one wait covers all
      size_t row = base + t0 + j;
      dtv[j] = dtW[row * DI_ + d];
      xv[j]  = (float)x_bf[row * DI_ + d];
      Bv[j]  = dbc[row * 96 + R_ + n];
      Cv[j]  = dbc[row * 96 + R_ + N_ + n];
    }
#pragma unroll
    for (int j = 0; j < SCH; ++j) {
      float dA = __expf(dtv[j] * Aval);      // 1 exp on the critical path
      st = dA * st + (dtv[j] * Bv[j]) * xv[j];
    }
    if (t0 + SCH == L_) { lastC = Cv[SCH - 1]; lastX = xv[SCH - 1]; }
  }
  float contrib = st * lastC;
#pragma unroll
  for (int m = 8; m >= 1; m >>= 1)
    contrib += __shfl_xor(contrib, m, 32);   // reduce over n (16-lane groups)
  if (n == 0) y_last[b * DI_ + d] = contrib + lastX * Dp[d];
}

// ---- z for last token only + SiLU gate -------------------------------------
__global__ __launch_bounds__(256)
void k_zgate(const __bf16* __restrict__ h_bf, const float* __restrict__ w_in,
             float* __restrict__ y_last) {
  int b = blockIdx.x >> 3;
  int d = ((blockIdx.x & 7) << 8) + threadIdx.x;
  __shared__ float sh[D_];
  const __bf16* hrow = h_bf + (size_t)(b * L_ + (L_ - 1)) * D_;
  for (int i = threadIdx.x; i < D_; i += 256) sh[i] = (float)hrow[i];
  __syncthreads();
  const float* w = w_in + (size_t)(DI_ + d) * D_;
  float acc = 0.0f;
  for (int k = 0; k < D_; k += 4) {
    v4f wv = *(const v4f*)(w + k);
    acc += sh[k] * wv[0] + sh[k + 1] * wv[1] + sh[k + 2] * wv[2] + sh[k + 3] * wv[3];
  }
  y_last[b * DI_ + d] *= acc * sigmoidf_(acc);
}

// ---- out-proj for last token: out_d[B,D] = y_last @ w_out^T -----------------
__global__ __launch_bounds__(256)
void k_outproj(const float* __restrict__ y_last, const float* __restrict__ w_out,
               float* __restrict__ out_d) {
  int b = blockIdx.x >> 2;
  int i = ((blockIdx.x & 3) << 8) + threadIdx.x;
  __shared__ float sy[DI_];
  for (int j = threadIdx.x; j < DI_; j += 256) sy[j] = y_last[b * DI_ + j];
  __syncthreads();
  const float* w = w_out + (size_t)i * DI_;
  float acc = 0.0f;
  for (int k = 0; k < DI_; k += 4) {
    v4f wv = *(const v4f*)(w + k);
    acc += sy[k] * wv[0] + sy[k + 1] * wv[1] + sy[k + 2] * wv[2] + sy[k + 3] * wv[3];
  }
  out_d[b * D_ + i] = acc;
}

// ---- head: logits[B,V] = out_d @ w_head^T + b_head (HBM streaming, NT) ------
__global__ __launch_bounds__(256)
void k_head(const float* __restrict__ out_d, const float* __restrict__ w_head,
            const float* __restrict__ b_head, float* __restrict__ out) {
  __shared__ float so[B__ * D_];
  for (int j = threadIdx.x; j < B__ * D_; j += 256) so[j] = out_d[j];
  __syncthreads();
  int v = blockIdx.x * 256 + threadIdx.x;
  if (v >= V_) return;
  const v4f* w = (const v4f*)(w_head + (size_t)v * D_);
  float a0 = 0.f, a1 = 0.f, a2 = 0.f, a3 = 0.f;
  for (int k4 = 0; k4 < D_ / 4; ++k4) {
    if ((k4 & 15) == 0) __builtin_prefetch((const void*)(w + k4 + 32), 0, 0);
    v4f wv = __builtin_nontemporal_load(w + k4);
    int k = k4 * 4;
    a0 += so[0 * D_ + k] * wv[0] + so[0 * D_ + k + 1] * wv[1] + so[0 * D_ + k + 2] * wv[2] + so[0 * D_ + k + 3] * wv[3];
    a1 += so[1 * D_ + k] * wv[0] + so[1 * D_ + k + 1] * wv[1] + so[1 * D_ + k + 2] * wv[2] + so[1 * D_ + k + 3] * wv[3];
    a2 += so[2 * D_ + k] * wv[0] + so[2 * D_ + k + 1] * wv[1] + so[2 * D_ + k + 2] * wv[2] + so[2 * D_ + k + 3] * wv[3];
    a3 += so[3 * D_ + k] * wv[0] + so[3 * D_ + k + 1] * wv[1] + so[3 * D_ + k + 2] * wv[2] + so[3 * D_ + k + 3] * wv[3];
  }
  float bb = b_head[v];
  out[0 * V_ + v] = a0 + bb;
  out[1 * V_ + v] = a1 + bb;
  out[2 * V_ + v] = a2 + bb;
  out[3 * V_ + v] = a3 + bb;
}

// ---- launcher ---------------------------------------------------------------
extern "C" void kernel_launch(void* const* d_in, const int* in_sizes, int n_in,
                              void* d_out, int out_size, void* d_ws, size_t ws_size,
                              hipStream_t stream) {
  const int*   tokens  = (const int*)  d_in[0];
  const float* emb     = (const float*)d_in[1];
  const float* w_in    = (const float*)d_in[2];
  const float* conv_w  = (const float*)d_in[3];
  const float* conv_b  = (const float*)d_in[4];
  const float* w_x     = (const float*)d_in[5];
  const float* w_dt    = (const float*)d_in[6];
  const float* b_dt    = (const float*)d_in[7];
  const float* A_log   = (const float*)d_in[8];
  const float* D_param = (const float*)d_in[9];
  const float* w_out   = (const float*)d_in[10];
  const float* w_head  = (const float*)d_in[11];
  const float* b_head  = (const float*)d_in[12];
  float* out = (float*)d_out;

  char* ws = (char*)d_ws;
  size_t off = 0;
  auto alloc = [&](size_t bytes) { char* p = ws + off; off = (off + bytes + 255) & ~(size_t)255; return p; };

  __bf16* h_bf      = (__bf16*)alloc((size_t)T_  * D_  * 2);
  __bf16* w_in_x_bf = (__bf16*)alloc((size_t)DI_ * D_  * 2);
  __bf16* w_x_bf    = (__bf16*)alloc((size_t)96  * DI_ * 2);
  __bf16* w_dt_bf   = (__bf16*)alloc((size_t)DI_ * R_  * 2);
  float*  x_pre     = (float*) alloc((size_t)T_  * DI_ * 4);
  __bf16* x_bf      = (__bf16*)alloc((size_t)T_  * DI_ * 2);
  float*  dbc       = (float*) alloc((size_t)T_  * 96  * 4);
  __bf16* dtr_bf    = (__bf16*)alloc((size_t)T_  * R_  * 2);
  float*  dtW       = (float*) alloc((size_t)T_  * DI_ * 4);
  float*  y_last    = (float*) alloc((size_t)B__ * DI_ * 4);
  float*  out_d     = (float*) alloc((size_t)B__ * D_  * 4);

  k_gather_bf16<<<T_, 256, 0, stream>>>(tokens, emb, h_bf);
  k_cast_bf16<<<(DI_ * D_ + 255) / 256, 256, 0, stream>>>(w_in, w_in_x_bf, DI_ * D_);
  k_cast_bf16<<<(96 * DI_ + 255) / 256, 256, 0, stream>>>(w_x, w_x_bf, 96 * DI_);
  k_cast_bf16<<<(DI_ * R_ + 255) / 256, 256, 0, stream>>>(w_dt, w_dt_bf, DI_ * R_);

  k_gemm_xpre<<<dim3(T_ / 128, DI_ / 64), 256, 0, stream>>>(h_bf, w_in_x_bf, x_pre);
  k_conv_silu<<<(T_ / TCH * DI_) / 256, 256, 0, stream>>>(x_pre, conv_w, conv_b, x_bf);
  k_gemm_dbc<<<dim3(T_ / 128, 1), 256, 0, stream>>>(x_bf, w_x_bf, dbc, dtr_bf);
  k_gemm_dt<<<dim3(T_ / 128, DI_ / 64), 256, 0, stream>>>(dtr_bf, w_dt_bf, b_dt, dtW);
  k_scan<<<B__ * (DI_ / 16), 256, 0, stream>>>(dtW, x_bf, dbc, A_log, D_param, y_last);
  k_zgate<<<B__ * (DI_ / 256), 256, 0, stream>>>(h_bf, w_in, y_last);
  k_outproj<<<B__ * (D_ / 256), 256, 0, stream>>>(y_last, w_out, out_d);
  k_head<<<(V_ + 255) / 256, 256, 0, stream>>>(out_d, w_head, b_head, out);
}
